// FrameTransformer_18648747999560
// MI455X (gfx1250) — compile-verified, matrix operand order checked by reading, not run on hardware
//
#include <hip/hip_runtime.h>
#include <math.h>

typedef __attribute__((ext_vector_type(16))) _Float16 v16h;
typedef __attribute__((ext_vector_type(8)))  float    v8f;

static constexpr int  FEAT   = 512;
static constexpr int  FRAMES = 512;
static constexpr long SP     = (long)FEAT * FRAMES; // 262144 elems per channel plane
static constexpr int  NB     = 2;                   // batch
static constexpr int  HEADS  = 4;
static constexpr int  DH     = 128;

// ------------------------------------------------------------------
// helpers
// ------------------------------------------------------------------
__device__ __forceinline__ float gelu_exact(float x) {
  return 0.5f * x * (1.0f + erff(x * 0.70710678118654752440f));
}

// 16x32 f16 operand fragment; rows are k-contiguous (leading dim = K).
// Per CDNA5 ISA 7.12.2: lane = hb*16 + r ; element e -> k = kk + (e&7) + (e>=8 ? 16 : 0)
// with kk = k0 + hb*8. Two 16B-aligned b128 loads per lane.
__device__ __forceinline__ v16h ldfrag(const _Float16* __restrict__ p) {
  union { v16h v; uint4 q[2]; } u;
  u.q[0] = *(const uint4*)(p);
  u.q[1] = *(const uint4*)(p + 16);
  return u.v;
}

// ------------------------------------------------------------------
// Batched GEMM, f16 inputs / f32 accumulate via v_wmma_f32_16x16x32_f16.
//   A: [m][K] packed f16, base = (z/aDiv)*aSo + (z%aDiv)*aSi
//   B: [n][K] packed f16 (i.e. B^T row-major), base = z*bSo
//   C: f32 or f16, elem addr = (z/cDiv)*cSo + (z%cDiv)*cSi + m*Csr + n*Csc
//   C = act( alpha*A@B^T (+ C if accum) )
// Block = 4 waves; block tile 64(M) x 128(N); wave tile 32x64.
// Requires M%64==0, N%128==0, K%32==0 (all call sites satisfy this).
// ------------------------------------------------------------------
__global__ void __launch_bounds__(128)
gemm_wmma_kernel(const _Float16* __restrict__ A, const _Float16* __restrict__ B,
                 float* __restrict__ Cf, _Float16* __restrict__ Ch,
                 int M, int N, int K,
                 int aDiv, long aSo, long aSi, long bSo,
                 int cDiv, long cSo, long cSi, long Csr, long Csc,
                 float alpha, int accum, int act, int outHalf) {
  const int z = blockIdx.z;
  const _Float16* Ab = A + (long)(z / aDiv) * aSo + (long)(z % aDiv) * aSi;
  const _Float16* Bb = B + (long)z * bSo;
  const long cBase = (long)(z / cDiv) * cSo + (long)(z % cDiv) * cSi;

  const int wave = threadIdx.x >> 5;
  const int lane = threadIdx.x & 31;
  const int mf = lane & 15;
  const int hb = lane >> 4;
  const long tM = (long)blockIdx.y * 64 + (long)(wave >> 1) * 32;
  const long tN = (long)blockIdx.x * 128 + (long)(wave & 1) * 64;

  const _Float16* Ar0 = Ab + (tM + mf) * (long)K;
  const _Float16* Ar1 = Ab + (tM + 16 + mf) * (long)K;
  const _Float16* Br0 = Bb + (tN + mf) * (long)K;
  const _Float16* Br1 = Bb + (tN + 16 + mf) * (long)K;
  const _Float16* Br2 = Bb + (tN + 32 + mf) * (long)K;
  const _Float16* Br3 = Bb + (tN + 48 + mf) * (long)K;

  v8f acc[2][4] = {};
  for (int k0 = 0; k0 < K; k0 += 32) {
    const int kk = k0 + hb * 8;
    v16h a0 = ldfrag(Ar0 + kk);
    v16h a1 = ldfrag(Ar1 + kk);
    v16h b0 = ldfrag(Br0 + kk);
    v16h b1 = ldfrag(Br1 + kk);
    v16h b2 = ldfrag(Br2 + kk);
    v16h b3 = ldfrag(Br3 + kk);
    acc[0][0] = __builtin_amdgcn_wmma_f32_16x16x32_f16(false, a0, false, b0, (short)0, acc[0][0], false, false);
    acc[0][1] = __builtin_amdgcn_wmma_f32_16x16x32_f16(false, a0, false, b1, (short)0, acc[0][1], false, false);
    acc[0][2] = __builtin_amdgcn_wmma_f32_16x16x32_f16(false, a0, false, b2, (short)0, acc[0][2], false, false);
    acc[0][3] = __builtin_amdgcn_wmma_f32_16x16x32_f16(false, a0, false, b3, (short)0, acc[0][3], false, false);
    acc[1][0] = __builtin_amdgcn_wmma_f32_16x16x32_f16(false, a1, false, b0, (short)0, acc[1][0], false, false);
    acc[1][1] = __builtin_amdgcn_wmma_f32_16x16x32_f16(false, a1, false, b1, (short)0, acc[1][1], false, false);
    acc[1][2] = __builtin_amdgcn_wmma_f32_16x16x32_f16(false, a1, false, b2, (short)0, acc[1][2], false, false);
    acc[1][3] = __builtin_amdgcn_wmma_f32_16x16x32_f16(false, a1, false, b3, (short)0, acc[1][3], false, false);
  }

  // C/D layout: v8f element vv -> m = vv + 8*hb, n = lane&15
#pragma unroll
  for (int i = 0; i < 2; ++i)
#pragma unroll
    for (int j = 0; j < 4; ++j)
#pragma unroll
      for (int vv = 0; vv < 8; ++vv) {
        long m = tM + i * 16 + (long)(hb * 8 + vv);
        long n = tN + j * 16 + mf;
        long idx = cBase + m * Csr + n * Csc;
        float val = acc[i][j][vv] * alpha;
        if (accum) val += Cf[idx];
        if (act) val = gelu_exact(val);
        if (outHalf) Ch[idx] = (_Float16)val;
        else         Cf[idx] = val;
      }
}

// ------------------------------------------------------------------
// 3x3 same-pad conv over (FEAT, FRAMES) spatial dims, NCHW f32.
// grid: (SP/256, Cout, NB)
// ------------------------------------------------------------------
__global__ void conv3x3_kernel(const float* __restrict__ in, const float* __restrict__ wt,
                               float* __restrict__ out, int Cin, int Cout) {
  int idx = blockIdx.x * blockDim.x + threadIdx.x; // f*512 + w
  int f = idx >> 9, w = idx & 511;
  int co = blockIdx.y, b = blockIdx.z;
  const float* ip = in + (long)b * Cin * SP;
  float acc = 0.f;
  for (int ci = 0; ci < Cin; ++ci) {
    const float* cp = ip + (long)ci * SP;
    const float* wp = wt + ((long)co * Cin + ci) * 9;
#pragma unroll
    for (int df = -1; df <= 1; ++df) {
      int ff = f + df;
      if (ff < 0 || ff > 511) continue;
#pragma unroll
      for (int dw = -1; dw <= 1; ++dw) {
        int ww = w + dw;
        if (ww < 0 || ww > 511) continue;
        acc += cp[ff * 512 + ww] * wp[(df + 1) * 3 + (dw + 1)];
      }
    }
  }
  out[((long)b * Cout + co) * SP + idx] = acc;
}

// LayerNorm over FEATURES axis per (b,c,w). grid: (512/256, C, NB)
__global__ void layernorm_kernel(const float* __restrict__ x, const float* __restrict__ gamma,
                                 const float* __restrict__ beta, float* __restrict__ y, int C) {
  int w = blockIdx.x * blockDim.x + threadIdx.x;
  int c = blockIdx.y, b = blockIdx.z;
  const float* xp = x + ((long)b * C + c) * SP + w;
  float s = 0.f, s2 = 0.f;
  for (int f = 0; f < 512; ++f) { float v = xp[f * 512]; s += v; s2 += v * v; }
  float mu   = s * (1.f / 512.f);
  float var  = s2 * (1.f / 512.f) - mu * mu;
  float rstd = rsqrtf(var + 1e-5f);
  float* yp = y + ((long)b * C + c) * SP + w;
  const float* g  = gamma + c * 512;
  const float* be = beta + c * 512;
  for (int f = 0; f < 512; ++f)
    yp[f * 512] = (xp[f * 512] - mu) * rstd * g[f] + be[f];
}

// rotary + head reorder: conv[b][(h*128+d)][w] -> f16 q[(b*4+h)][w][d]
__global__ void rotary_kernel(const float* __restrict__ conv, _Float16* __restrict__ outq) {
  int idx = blockIdx.x * 256 + threadIdx.x; // 2^18 = B*H*512*64 pair-slots
  int i = idx & 63;
  int w = (idx >> 6) & 511;
  int h = (idx >> 15) & 3;
  int b = idx >> 17;
  float t1 = conv[(long)b * SP + (long)(h * 128 + 2 * i) * 512 + w];
  float t2 = conv[(long)b * SP + (long)(h * 128 + 2 * i + 1) * 512 + w];
  float inv = powf(10000.f, -(2.f * (float)i) / 128.f);
  float ang = (float)w * inv;
  float cs = cosf(ang), sn = sinf(ang);
  _Float16* o = outq + (((long)(b * 4 + h) * 512 + w) * 128);
  o[2 * i]     = (_Float16)(t1 * cs - t2 * sn);
  o[2 * i + 1] = (_Float16)(t1 * sn + t2 * cs);
}

// row softmax over 512 cols; 128 threads per row; f32 in, f16 out
__global__ void softmax_kernel(const float* __restrict__ in, _Float16* __restrict__ out) {
  long row = blockIdx.x;
  const float* r = in + row * 512;
  _Float16* o = out + row * 512;
  __shared__ float red[128];
  int t = threadIdx.x;
  float v[4];
#pragma unroll
  for (int j = 0; j < 4; ++j) v[j] = r[t + j * 128];
  float mx = fmaxf(fmaxf(v[0], v[1]), fmaxf(v[2], v[3]));
  red[t] = mx; __syncthreads();
  for (int s = 64; s > 0; s >>= 1) { if (t < s) red[t] = fmaxf(red[t], red[t + s]); __syncthreads(); }
  mx = red[0]; __syncthreads();
  float sum = 0.f;
#pragma unroll
  for (int j = 0; j < 4; ++j) { v[j] = expf(v[j] - mx); sum += v[j]; }
  red[t] = sum; __syncthreads();
  for (int s = 64; s > 0; s >>= 1) { if (t < s) red[t] += red[t + s]; __syncthreads(); }
  float inv = 1.0f / red[0];
#pragma unroll
  for (int j = 0; j < 4; ++j) o[t + j * 128] = (_Float16)(v[j] * inv);
}

__global__ void gelu_kernel(float* __restrict__ x, long n) {
  long idx = (long)blockIdx.x * blockDim.x + threadIdx.x;
  if (idx < n) x[idx] = gelu_exact(x[idx]);
}

// plain f32 -> f16 conversion
__global__ void cvt_f16_kernel(const float* __restrict__ in, _Float16* __restrict__ out, long n) {
  long idx = (long)blockIdx.x * blockDim.x + threadIdx.x;
  if (idx < n) out[idx] = (_Float16)in[idx];
}

// per-z 512x512 transpose with f32->f16 cvt (LDS tiled). block (32,8), grid (16,16,z)
__global__ void transpose_cvt_kernel(const float* __restrict__ in, _Float16* __restrict__ out) {
  __shared__ float tile[32][33];
  long zb = (long)blockIdx.z * SP;
  int bx = blockIdx.x * 32, by = blockIdx.y * 32;
  int tx = threadIdx.x, ty = threadIdx.y;
#pragma unroll
  for (int j = 0; j < 32; j += 8)
    tile[ty + j][tx] = in[zb + (long)(by + ty + j) * 512 + bx + tx];
  __syncthreads();
#pragma unroll
  for (int j = 0; j < 32; j += 8)
    out[zb + (long)(bx + ty + j) * 512 + by + tx] = (_Float16)tile[tx][ty + j];
}

// h[b,c,f,w] += res[c,f] * z[b,c,f,w]   (C==5)
__global__ void residual_kernel(float* __restrict__ h, const float* __restrict__ z,
                                const float* __restrict__ res, long total) {
  long idx = (long)blockIdx.x * blockDim.x + threadIdx.x;
  if (idx >= total) return;
  long s = idx % SP;
  int c = (int)((idx / SP) % 5);
  int f = (int)(s >> 9);
  h[idx] += res[c * 512 + f] * z[idx];
}

// channel-slice copy between NCHW tensors with different channel counts
__global__ void copy_chan_kernel(const float* __restrict__ src, float* __restrict__ dst,
                                 int srcC, int dstC, int srcOff, int dstOff, int n, long total) {
  long idx = (long)blockIdx.x * blockDim.x + threadIdx.x;
  if (idx >= total) return;
  long s = idx % SP;
  long t = idx / SP;
  int c = (int)(t % n);
  int b = (int)(t / n);
  dst[((long)b * dstC + dstOff + c) * SP + s] =
      src[((long)b * srcC + srcOff + c) * SP + s];
}

// ------------------------------------------------------------------
// host-side launch helpers
// ------------------------------------------------------------------
static void gemm(hipStream_t st, const _Float16* A, const _Float16* B, void* C,
                 int M, int N, int K, int batch,
                 int aDiv, long aSo, long aSi, long bSo,
                 int cDiv, long cSo, long cSi, long Csr, long Csc,
                 float alpha, int accum, int act, int outHalf) {
  dim3 g(N / 128, M / 64, batch);
  gemm_wmma_kernel<<<g, 128, 0, st>>>(A, B, (float*)C, (_Float16*)C, M, N, K,
                                      aDiv, aSo, aSi, bSo, cDiv, cSo, cSi, Csr, Csc,
                                      alpha, accum, act, outHalf);
}

static void copy_chan(hipStream_t st, const float* src, float* dst,
                      int srcC, int dstC, int srcOff, int dstOff, int n) {
  long total = (long)NB * n * SP;
  copy_chan_kernel<<<(unsigned)((total + 255) / 256), 256, 0, st>>>(
      src, dst, srcC, dstC, srcOff, dstOff, n, total);
}

static void cvt16(hipStream_t st, const float* src, _Float16* dst, long n) {
  cvt_f16_kernel<<<(unsigned)((n + 255) / 256), 256, 0, st>>>(src, dst, n);
}

extern "C" void kernel_launch(void* const* d_in, const int* in_sizes, int n_in,
                              void* d_out, int out_size, void* d_ws, size_t ws_size,
                              hipStream_t stream) {
  struct LayerP { const float *gamma,*beta,*res,*wc1,*wc2,*wl1,*wl2,*wk,*wo,*wq,*wv; };
  LayerP L[4];
  const float *x, *pe1, *pe2;

  if (n_in > 0 && in_sizes[0] == 2 * 2 * 512 * 512) {
    // insertion-order flatten: x, pe_w1, pe_w2, then per layer:
    // gamma, beta, wq, wk, wv, wo, w_conv1, w_lin1, w_lin2, w_conv2, res
    int i = 0;
    x   = (const float*)d_in[i++];
    pe1 = (const float*)d_in[i++];
    pe2 = (const float*)d_in[i++];
    for (int l = 0; l < 4; ++l) {
      L[l].gamma = (const float*)d_in[i++];
      L[l].beta  = (const float*)d_in[i++];
      L[l].wq    = (const float*)d_in[i++];
      L[l].wk    = (const float*)d_in[i++];
      L[l].wv    = (const float*)d_in[i++];
      L[l].wo    = (const float*)d_in[i++];
      L[l].wc1   = (const float*)d_in[i++];
      L[l].wl1   = (const float*)d_in[i++];
      L[l].wl2   = (const float*)d_in[i++];
      L[l].wc2   = (const float*)d_in[i++];
      L[l].res   = (const float*)d_in[i++];
    }
  } else {
    // jax tree_flatten (sorted keys): layers first, each layer dict sorted:
    // beta, gamma, res, w_conv1, w_conv2, w_lin1, w_lin2, wk, wo, wq, wv ; then pe_w1, pe_w2, x
    int i = 0;
    for (int l = 0; l < 4; ++l) {
      L[l].beta  = (const float*)d_in[i++];
      L[l].gamma = (const float*)d_in[i++];
      L[l].res   = (const float*)d_in[i++];
      L[l].wc1   = (const float*)d_in[i++];
      L[l].wc2   = (const float*)d_in[i++];
      L[l].wl1   = (const float*)d_in[i++];
      L[l].wl2   = (const float*)d_in[i++];
      L[l].wk    = (const float*)d_in[i++];
      L[l].wo    = (const float*)d_in[i++];
      L[l].wq    = (const float*)d_in[i++];
      L[l].wv    = (const float*)d_in[i++];
    }
    pe1 = (const float*)d_in[i++];
    pe2 = (const float*)d_in[i++];
    x   = (const float*)d_in[i++];
  }

  // ---- workspace layout ----
  float* ws = (float*)d_ws;
  long off = 0;
  auto allocF = [&](long n) { float* p = ws + off; off += n; return p; };
  auto allocH = [&](long n) { _Float16* p = (_Float16*)(ws + off); off += n / 2; return p; };

  // f32 buffers
  float* h     = allocF(NB * 5 * SP);     // running hidden state
  float* hn    = allocF(NB * 5 * SP);     // layernorm output
  float* cat   = allocF(NB * 9 * SP);     // concat buffer (hin / zin)
  float* qc    = allocF(NB * SP);         // conv outputs (feature-major)
  float* kc    = allocF(NB * SP);
  float* vc    = allocF(NB * SP);
  float* qk    = allocF(NB * HEADS * SP); // accumulated attention scores (f32)
  float* aout  = allocF(NB * SP);         // after wo (f32, feeds zin concat)
  float* pattn = allocF(NB * 4 * SP);     // accumulated attention channels
  float* z1    = allocF(NB * 12 * SP);    // conv1 out f32 [f][w]
  float* z3    = allocF(NB * 12 * SP);    // lin2 out f32 [f][w]
  float* z4    = allocF(NB * 5 * SP);     // conv2 out
  // f16 buffers (all GEMM operands; k-contiguous rows)
  _Float16* qb16   = allocH(NB * SP);        // [z=b*4+h][w][d]
  _Float16* kb16   = allocH(NB * SP);        // [z][u][d]
  _Float16* vt16   = allocH(NB * SP);        // [z][d][u]  (== conv layout, cvt only)
  _Float16* pr16   = allocH(NB * HEADS * SP);// [z][w][u]
  _Float16* aft16  = allocH(NB * SP);        // [b][w][f]  (pv output, wo's B operand)
  _Float16* z1t16  = allocH(NB * 12 * SP);   // [z][w][f]  (lin1 B operand)
  _Float16* z2t16  = allocH(NB * 24 * SP);   // [z][w][g=1024] (lin1 out, lin2 B operand)
  _Float16* wo16   = allocH(SP);             // per-layer converted weights
  _Float16* wl1_16 = allocH(12L * 1024 * 512);
  _Float16* wl2_16 = allocH(12L * 512 * 1024);

  float* pemid = z1;  // aliases: pe path runs before layers touch z1/z3
  float* pbuf  = z3;

  // ---- positional encoding path ----
  conv3x3_kernel<<<dim3(SP / 256, 8, NB), 256, 0, stream>>>(x, pe1, pemid, 2, 8);
  {
    long n = (long)NB * 8 * SP;
    gelu_kernel<<<(unsigned)((n + 255) / 256), 256, 0, stream>>>(pemid, n);
  }
  conv3x3_kernel<<<dim3(SP / 256, 1, NB), 256, 0, stream>>>(pemid, pe2, pbuf, 8, 1);
  // h = concat([x, x, p], axis=1)
  copy_chan(stream, x,    h, 2, 5, 0, 0, 2);
  copy_chan(stream, x,    h, 2, 5, 0, 2, 2);
  copy_chan(stream, pbuf, h, 1, 5, 0, 4, 1);

  const float inv_sqrt_f = 0.044194173824159216f; // 1/sqrt(512)

  for (int l = 0; l < 4; ++l) {
    int cinA = 5 + l;   // attention conv input channels
    int cinC = 6 + l;   // conv1 input channels

    // weights -> f16 (cheap, L2-resident)
    cvt16(stream, L[l].wo,  wo16,   SP);
    cvt16(stream, L[l].wl1, wl1_16, 12L * 1024 * 512);
    cvt16(stream, L[l].wl2, wl2_16, 12L * 512 * 1024);

    layernorm_kernel<<<dim3(2, 5, NB), 256, 0, stream>>>(h, L[l].gamma, L[l].beta, hn, 5);

    // hin = concat([hnorm, pattn])
    copy_chan(stream, hn, cat, 5, cinA, 0, 0, 5);
    if (l > 0) copy_chan(stream, pattn, cat, 4, cinA, 0, 5, l);

    conv3x3_kernel<<<dim3(SP / 256, 1, NB), 256, 0, stream>>>(cat, L[l].wq, qc, cinA, 1);
    conv3x3_kernel<<<dim3(SP / 256, 1, NB), 256, 0, stream>>>(cat, L[l].wk, kc, cinA, 1);
    conv3x3_kernel<<<dim3(SP / 256, 1, NB), 256, 0, stream>>>(cat, L[l].wv, vc, cinA, 1);

    rotary_kernel<<<1024, 256, 0, stream>>>(qc, qb16);
    rotary_kernel<<<1024, 256, 0, stream>>>(kc, kb16);
    cvt16(stream, vc, vt16, NB * SP);   // conv layout [b][h*128+d][u] == [z][d][u]

    // scores: qk[z] (+)= q[z] @ k[z]^T * 1/sqrt(F)    (z = b*4+h)
    gemm(stream, qb16, kb16, qk, 512, 512, DH, NB * HEADS,
         1, (long)512 * DH, 0, (long)512 * DH,
         1, SP, 0, 512, 1,
         inv_sqrt_f, (l > 0) ? 1 : 0, 0, 0);

    softmax_kernel<<<NB * HEADS * 512, 128, 0, stream>>>(qk, pr16);

    // pv: aft[b][w][h*128+d] = (probs[z] @ v[z]) [w,d]  -> f16, B-operand layout for wo
    gemm(stream, pr16, vt16, aft16, 512, DH, 512, NB * HEADS,
         1, SP, 0, (long)512 * DH,
         HEADS, SP, DH, 512, 1,
         1.0f, 0, 0, 1);

    // wo: aout[b][g][w] = wo @ aft[b]^T   (weight shared across batch)
    gemm(stream, wo16, aft16, aout, 512, 512, 512, NB,
         1, 0, 0, SP,
         1, SP, 0, 512, 1,
         1.0f, 0, 0, 0);

    // zin = concat([hnorm, aout, pattn])
    copy_chan(stream, hn,   cat, 5, cinC, 0, 0, 5);
    copy_chan(stream, aout, cat, 1, cinC, 0, 5, 1);
    if (l > 0) copy_chan(stream, pattn, cat, 4, cinC, 0, 6, l);

    conv3x3_kernel<<<dim3(SP / 256, 12, NB), 256, 0, stream>>>(cat, L[l].wc1, z1, cinC, 12);
    // z1 [z][f][w] f32 -> z1t [z][w][f] f16 (LDS tiled transpose)
    transpose_cvt_kernel<<<dim3(16, 16, NB * 12), dim3(32, 8), 0, stream>>>(z1, z1t16);

    // lin1 + GELU: z2t[z][w][g] = gelu(w_lin1[c] @ z1[z]) (transposed f16 store)
    gemm(stream, wl1_16, z1t16, z2t16, 1024, 512, 512, NB * 12,
         12, 0, (long)1024 * 512, SP,
         1, (long)512 * 1024, 0, 1, 1024,
         1.0f, 0, 1, 1);

    // lin2: z3[z][f][w] = w_lin2[c] @ z2[z]
    gemm(stream, wl2_16, z2t16, z3, 512, 512, 1024, NB * 12,
         12, 0, (long)512 * 1024, (long)512 * 1024,
         1, SP, 0, 512, 1,
         1.0f, 0, 0, 0);

    conv3x3_kernel<<<dim3(SP / 256, 5, NB), 256, 0, stream>>>(z3, L[l].wc2, z4, 12, 5);

    {
      long n = (long)NB * 5 * SP;
      residual_kernel<<<(unsigned)((n + 255) / 256), 256, 0, stream>>>(h, z4, L[l].res, n);
    }

    // new_attn = concat([pattn, aout]) -> append aout at channel l
    copy_chan(stream, aout, pattn, 1, 4, 0, l, 1);
  }

  // output = h[:, :2]
  copy_chan(stream, h, (float*)d_out, 5, 2, 0, 0, 2);
  (void)in_sizes; (void)n_in; (void)out_size; (void)ws_size;
}
